// BasicBiPointNetSemSeg_63007170232712
// MI455X (gfx1250) — 8-bit matrix operand order repaired and checked by reading, compiled, not run on hardware
//
#include <hip/hip_runtime.h>
#include <cstdint>


#define PTS    65536
#define NBATCH 16
#define NCLS   13

typedef __attribute__((ext_vector_type(8))) int v8i;

#define SGB_VMEM_READ 0x020
#define SGB_MFMA      0x008

enum { EPI_SIGN = 0, EPI_FLOAT_HT = 1, EPI_BOTH = 2, EPI_MAXPOOL = 3, EPI_RAW = 4 };

__device__ __forceinline__ int sgn8(float v) { return v > 0.f ? 1 : (v < 0.f ? -1 : 0); }

__device__ __forceinline__ unsigned ordf(float f) {
  unsigned u = __float_as_uint(f);
  return (u & 0x80000000u) ? ~u : (u | 0x80000000u);
}
__device__ __forceinline__ float unordf(unsigned o) {
  unsigned u = (o & 0x80000000u) ? (o ^ 0x80000000u) : ~o;
  return __uint_as_float(u);
}

__device__ __forceinline__ void load_frags(
    const signed char* __restrict__ arow,
    const signed char* __restrict__ w0, const signed char* __restrict__ w1,
    const signed char* __restrict__ w2, const signed char* __restrict__ w3,
    int k, v8i& a, v8i& b0, v8i& b1, v8i& b2, v8i& b3) {
#pragma unroll
  for (int i = 0; i < 8; ++i)
    a[i] = *(const int*)(arow + k + 32 * (i >> 2) + 4 * (i & 3));
#pragma unroll
  for (int j = 0; j < 8; ++j) {
    const int koff = k + 32 * (j >> 2) + 4 * (j & 3);
    b0[j] = *(const int*)(w0 + koff);
    b1[j] = *(const int*)(w1 + koff);
    b2[j] = *(const int*)(w2 + koff);
    b3[j] = *(const int*)(w3 + koff);
  }
}

template <int EPI, bool HASBN>
__global__ __launch_bounds__(128) void bilinear_wmma_kernel(
    const signed char* __restrict__ A, int strideA,
    const signed char* __restrict__ Wb, int Kpad,
    const float* __restrict__ bias,
    const float* __restrict__ bn_be, const float* __restrict__ bn_g,
    const float* __restrict__ bn_m,  const float* __restrict__ bn_v,
    int M, int Cout,
    signed char* __restrict__ outS, int strideS,
    float* __restrict__ outF, int strideF,
    unsigned* __restrict__ pool)
{
  const int wave = threadIdx.x >> 5;
  const int lane = threadIdx.x & 31;
  const int m0 = blockIdx.y * 64 + wave * 16;
  if (m0 >= M) return;
  const int n0 = blockIdx.x * 64;
  const int ml   = lane & 15;
  const int half = lane >> 4;

  v8i acc0 = {}, acc1 = {}, acc2 = {}, acc3 = {};
  const signed char* arow  = A + (size_t)(m0 + ml) * strideA + 16 * half;
  const signed char* wrow0 = Wb + (size_t)(n0 + 0  + ml) * Kpad + 16 * half;
  const signed char* wrow1 = Wb + (size_t)(n0 + 16 + ml) * Kpad + 16 * half;
  const signed char* wrow2 = Wb + (size_t)(n0 + 32 + ml) * Kpad + 16 * half;
  const signed char* wrow3 = Wb + (size_t)(n0 + 48 + ml) * Kpad + 16 * half;

  int k = 0;
  for (; k + 128 <= Kpad; k += 128) {
    v8i a0, p0, p1, p2, p3;
    load_frags(arow, wrow0, wrow1, wrow2, wrow3, k, a0, p0, p1, p2, p3);
    v8i a1, q0, q1, q2, q3;
    load_frags(arow, wrow0, wrow1, wrow2, wrow3, k + 64, a1, q0, q1, q2, q3);
    acc0 = __builtin_amdgcn_wmma_i32_16x16x64_iu8(true, a0, true, p0, acc0, false, false);
    acc1 = __builtin_amdgcn_wmma_i32_16x16x64_iu8(true, a0, true, p1, acc1, false, false);
    acc2 = __builtin_amdgcn_wmma_i32_16x16x64_iu8(true, a0, true, p2, acc2, false, false);
    acc3 = __builtin_amdgcn_wmma_i32_16x16x64_iu8(true, a0, true, p3, acc3, false, false);
    acc0 = __builtin_amdgcn_wmma_i32_16x16x64_iu8(true, a1, true, q0, acc0, false, false);
    acc1 = __builtin_amdgcn_wmma_i32_16x16x64_iu8(true, a1, true, q1, acc1, false, false);
    acc2 = __builtin_amdgcn_wmma_i32_16x16x64_iu8(true, a1, true, q2, acc2, false, false);
    acc3 = __builtin_amdgcn_wmma_i32_16x16x64_iu8(true, a1, true, q3, acc3, false, false);
    __builtin_amdgcn_sched_group_barrier(SGB_VMEM_READ, 12, 0);
    __builtin_amdgcn_sched_group_barrier(SGB_VMEM_READ, 12, 0);
    __builtin_amdgcn_sched_group_barrier(SGB_MFMA, 4, 0);
    __builtin_amdgcn_sched_group_barrier(SGB_MFMA, 4, 0);
  }
  if (k < Kpad) {
    v8i a0, p0, p1, p2, p3;
    load_frags(arow, wrow0, wrow1, wrow2, wrow3, k, a0, p0, p1, p2, p3);
    acc0 = __builtin_amdgcn_wmma_i32_16x16x64_iu8(true, a0, true, p0, acc0, false, false);
    acc1 = __builtin_amdgcn_wmma_i32_16x16x64_iu8(true, a0, true, p1, acc1, false, false);
    acc2 = __builtin_amdgcn_wmma_i32_16x16x64_iu8(true, a0, true, p2, acc2, false, false);
    acc3 = __builtin_amdgcn_wmma_i32_16x16x64_iu8(true, a0, true, p3, acc3, false, false);
    __builtin_amdgcn_sched_group_barrier(SGB_VMEM_READ, 12, 0);
    __builtin_amdgcn_sched_group_barrier(SGB_MFMA, 4, 0);
  }

  const v8i accs[4] = {acc0, acc1, acc2, acc3};
#pragma unroll
  for (int t = 0; t < 4; ++t) {
    const int n = n0 + t * 16 + ml;
    const bool nok = (n < Cout);
    float bi = nok ? bias[n] : 0.f;
    float sc = 1.f, mu = 0.f, be = 0.f;
    if (HASBN && nok) {
      sc = bn_g[n] * rsqrtf(bn_v[n] + 1e-5f);
      mu = bn_m[n];
      be = bn_be[n];
    }
    float vmax = -3.4e38f;
#pragma unroll
    for (int r = 0; r < 8; ++r) {
      const int row = m0 + r + 8 * half;
      float val = (float)accs[t][r] + bi;
      if (HASBN) val = (val - mu) * sc + be;
      if (EPI == EPI_FLOAT_HT || EPI == EPI_BOTH) {
        float fv = fminf(fmaxf(val, -1.f), 1.f);
        if (nok) outF[(size_t)row * strideF + n] = fv;
      }
      if (EPI == EPI_SIGN || EPI == EPI_BOTH) {
        if (nok) outS[(size_t)row * strideS + n] = (signed char)sgn8(val);
      }
      if (EPI == EPI_RAW) {
        if (nok) outF[(size_t)row * strideF + n] = val;
      }
      if (EPI == EPI_MAXPOOL) vmax = fmaxf(vmax, val);
    }
    if (EPI == EPI_MAXPOOL) {
      float other = __shfl_xor(vmax, 16, 32);
      vmax = fmaxf(vmax, other);
      if (half == 0 && nok) {
        int b = m0 >> 12;
        atomicMax(&pool[(size_t)b * Cout + n], ordf(vmax));
      }
    }
  }
}

__global__ void binarize_w_kernel(const float* __restrict__ W,
                                  signed char* __restrict__ out,
                                  int Cout, int Cin, int Npad, int Kpad) {
  int idx = blockIdx.x * blockDim.x + threadIdx.x;
  int ndw = Npad * (Kpad >> 2);
  if (idx >= ndw) return;
  int r  = idx / (Kpad >> 2);
  int kd = (idx - r * (Kpad >> 2)) * 4;
  int word = 0;
#pragma unroll
  for (int j = 0; j < 4; ++j) {
    int k = kd + j, s = 0;
    if (r < Cout && k < Cin) s = sgn8(W[(size_t)r * Cin + k]);
    word |= (s & 0xFF) << (8 * j);
  }
  *(int*)(out + (size_t)r * Kpad + kd) = word;
}

__global__ void fill_u32(unsigned* p, unsigned v, int n) {
  int i = blockIdx.x * blockDim.x + threadIdx.x;
  if (i < n) p[i] = v;
}

__global__ void build_sign_x9(const float* __restrict__ pos,
                              const float* __restrict__ feat,
                              signed char* __restrict__ out) {
  int p = blockIdx.x * blockDim.x + threadIdx.x;
  if (p >= PTS) return;
  const float* ps = pos + (size_t)p * 3;
  const float* f  = feat + (size_t)p * 6;
  int w0 = (sgn8(ps[0]) & 0xff) | ((sgn8(ps[1]) & 0xff) << 8) |
           ((sgn8(ps[2]) & 0xff) << 16) | ((sgn8(f[0]) & 0xff) << 24);
  int w1 = (sgn8(f[1]) & 0xff) | ((sgn8(f[2]) & 0xff) << 8) |
           ((sgn8(f[3]) & 0xff) << 16) | ((sgn8(f[4]) & 0xff) << 24);
  int w2 = (sgn8(f[5]) & 0xff);
  int* dst = (int*)(out + (size_t)p * 64);
  dst[0] = w0; dst[1] = w1; dst[2] = w2;
#pragma unroll
  for (int i = 3; i < 16; ++i) dst[i] = 0;
}

__global__ void add_eye_kernel(float* t, int k) {
  int i = blockIdx.x * blockDim.x + threadIdx.x;
  if (i >= NBATCH * k) return;
  int b = i / k, d = i - b * k;
  t[(size_t)b * k * k + d * (k + 1)] += 1.0f;
}

__global__ void einsum_xyz_kernel(const float* __restrict__ pos,
                                  const float* __restrict__ feat,
                                  const float* __restrict__ trans,
                                  signed char* __restrict__ out) {
  int p = blockIdx.x * blockDim.x + threadIdx.x;
  if (p >= PTS) return;
  int b = p >> 12;
  const float* T = trans + b * 9;
  float x = pos[(size_t)p * 3 + 0], y = pos[(size_t)p * 3 + 1], z = pos[(size_t)p * 3 + 2];
  float o0 = x * T[0] + y * T[3] + z * T[6];
  float o1 = x * T[1] + y * T[4] + z * T[7];
  float o2 = x * T[2] + y * T[5] + z * T[8];
  const float* f = feat + (size_t)p * 6;
  int w0 = (sgn8(o0) & 0xff) | ((sgn8(o1) & 0xff) << 8) |
           ((sgn8(o2) & 0xff) << 16) | ((sgn8(f[0]) & 0xff) << 24);
  int w1 = (sgn8(f[1]) & 0xff) | ((sgn8(f[2]) & 0xff) << 8) |
           ((sgn8(f[3]) & 0xff) << 16) | ((sgn8(f[4]) & 0xff) << 24);
  int w2 = (sgn8(f[5]) & 0xff);
  int* dst = (int*)(out + (size_t)p * 64);
  dst[0] = w0; dst[1] = w1; dst[2] = w2;
#pragma unroll
  for (int i = 3; i < 16; ++i) dst[i] = 0;
}

__global__ void einsum_feat_kernel(const float* __restrict__ f64,
                                   const float* __restrict__ tf,
                                   signed char* __restrict__ hf) {
  int j = threadIdx.x & 63;
  int p = blockIdx.x * 4 + (threadIdx.x >> 6);
  if (p >= PTS) return;
  int b = p >> 12;
  const float* xr = f64 + (size_t)p * 64;
  const float* T  = tf + (size_t)b * 4096;
  float s = 0.f;
#pragma unroll 8
  for (int i = 0; i < 64; ++i) s += xr[i] * T[i * 64 + j];
  hf[(size_t)p * 1088 + 1024 + j] = (signed char)sgn8(s);
}

__global__ void pool_sign_kernel(const unsigned* __restrict__ pool,
                                 signed char* __restrict__ dst, int n) {
  int i = blockIdx.x * blockDim.x + threadIdx.x;
  if (i < n) dst[i] = (signed char)sgn8(unordf(pool[i]));
}

__global__ void bcast_global_sign(const signed char* __restrict__ sp,
                                  signed char* __restrict__ hf) {
  int idx = blockIdx.x * blockDim.x + threadIdx.x;
  int p = idx >> 8, d = idx & 255;
  if (p >= PTS) return;
  int b = p >> 12;
  *(int*)(hf + (size_t)p * 1088 + d * 4) =
      *(const int*)(sp + (size_t)b * 1024 + d * 4);
}

__global__ void head_final_kernel(const float* __restrict__ h3,
                                  const float* __restrict__ W,
                                  const float* __restrict__ bias,
                                  float* __restrict__ out) {
  __shared__ float sw[NCLS * 128];
  __shared__ float sb[NCLS];
  for (int i = threadIdx.x; i < NCLS * 128; i += blockDim.x) sw[i] = W[i];
  if (threadIdx.x < NCLS) sb[threadIdx.x] = bias[threadIdx.x];
  __syncthreads();
  int p = blockIdx.x * blockDim.x + threadIdx.x;
  if (p >= PTS) return;
  const float* xr = h3 + (size_t)p * 128;
  float lg[NCLS];
#pragma unroll
  for (int c = 0; c < NCLS; ++c) lg[c] = sb[c];
  for (int i = 0; i < 128; ++i) {
    float xi = xr[i];
#pragma unroll
    for (int c = 0; c < NCLS; ++c) lg[c] += xi * sw[c * 128 + i];
  }
  float mx = lg[0];
#pragma unroll
  for (int c = 1; c < NCLS; ++c) mx = fmaxf(mx, lg[c]);
  float lse = 0.f;
#pragma unroll
  for (int c = 0; c < NCLS; ++c) lse += __expf(lg[c] - mx);
  lse = __logf(lse);
#pragma unroll
  for (int c = 0; c < NCLS; ++c) out[(size_t)p * NCLS + c] = lg[c] - mx - lse;
}

extern "C" void kernel_launch(void* const* d_in, const int* in_sizes, int n_in,
                              void* d_out, int out_size, void* d_ws, size_t ws_size,
                              hipStream_t stream) {
  (void)in_sizes; (void)n_in; (void)out_size; (void)ws_size;
  const float* pos  = (const float*)d_in[0];
  const float* feat = (const float*)d_in[1];
  float* out   = (float*)d_out;
  float* trans = out + (size_t)PTS * NCLS;
  float* tfeat = trans + NBATCH * 9;

  char* ws = (char*)d_ws;
  size_t off = 0;
  auto alloc = [&](size_t bytes) -> void* {
    void* p = ws + off;
    off += (bytes + 255) & ~(size_t)255;
    return p;
  };

  struct BW { int idx, co, ci; signed char* p; int npad, kpad; };
  BW bw[18] = {
      {92, 64, 9, nullptr, 0, 0},  {94, 128, 64, nullptr, 0, 0},  {96, 1024, 128, nullptr, 0, 0},
      {98, 512, 1024, nullptr, 0, 0}, {100, 256, 512, nullptr, 0, 0}, {102, 9, 256, nullptr, 0, 0},
      {60, 64, 64, nullptr, 0, 0}, {62, 128, 64, nullptr, 0, 0},  {64, 1024, 128, nullptr, 0, 0},
      {66, 512, 1024, nullptr, 0, 0}, {68, 256, 512, nullptr, 0, 0}, {70, 4096, 256, nullptr, 0, 0},
      {34, 64, 9, nullptr, 0, 0},  {36, 128, 64, nullptr, 0, 0},  {38, 1024, 128, nullptr, 0, 0},
      {14, 512, 1088, nullptr, 0, 0}, {16, 256, 512, nullptr, 0, 0}, {18, 128, 256, nullptr, 0, 0},
  };
  for (int i = 0; i < 18; ++i) {
    bw[i].npad = (bw[i].co + 63) & ~63;
    bw[i].kpad = (bw[i].ci + 63) & ~63;
    bw[i].p = (signed char*)alloc((size_t)bw[i].npad * bw[i].kpad);
    int ndw = bw[i].npad * (bw[i].kpad >> 2);
    binarize_w_kernel<<<(ndw + 255) / 256, 256, 0, stream>>>(
        (const float*)d_in[bw[i].idx], bw[i].p, bw[i].co, bw[i].ci, bw[i].npad, bw[i].kpad);
  }

  signed char* R1 = (signed char*)alloc((size_t)PTS * 512);
  signed char* R2 = (signed char*)alloc((size_t)PTS * 512);
  signed char* HF = (signed char*)alloc((size_t)PTS * 1088);
  float* F64 = (float*)alloc((size_t)PTS * 64 * sizeof(float));
  float* H3F = (float*)alloc((size_t)PTS * 128 * sizeof(float));
  unsigned* pools = (unsigned*)alloc(3 * 16 * 1024 * sizeof(unsigned));
  unsigned *pool1 = pools, *pool2 = pools + 16 * 1024, *pool3 = pools + 32 * 1024;
  signed char* fcA = (signed char*)alloc(16 * 1024);
  signed char* fcB = (signed char*)alloc(16 * 1024);
  signed char* sp3 = (signed char*)alloc(16 * 1024);

  auto gemm = [&](const signed char* A, int strideA, const BW& w,
                  int biasIdx, int bnBase, int M, int epi,
                  signed char* outS, int strideS, float* outF, int strideF,
                  unsigned* pool) {
    const float *be = nullptr, *g = nullptr, *m = nullptr, *v = nullptr;
    if (bnBase >= 0) {
      be = (const float*)d_in[bnBase + 0];
      g  = (const float*)d_in[bnBase + 1];
      m  = (const float*)d_in[bnBase + 2];
      v  = (const float*)d_in[bnBase + 3];
    }
    const float* bias = (const float*)d_in[biasIdx];
    dim3 grid((w.co + 63) / 64, (M + 63) / 64);
    dim3 blk(128);
    switch (epi) {
      case EPI_SIGN:
        bilinear_wmma_kernel<EPI_SIGN, true><<<grid, blk, 0, stream>>>(
            A, strideA, w.p, w.kpad, bias, be, g, m, v, M, w.co,
            outS, strideS, outF, strideF, pool);
        break;
      case EPI_FLOAT_HT:
        bilinear_wmma_kernel<EPI_FLOAT_HT, true><<<grid, blk, 0, stream>>>(
            A, strideA, w.p, w.kpad, bias, be, g, m, v, M, w.co,
            outS, strideS, outF, strideF, pool);
        break;
      case EPI_BOTH:
        bilinear_wmma_kernel<EPI_BOTH, true><<<grid, blk, 0, stream>>>(
            A, strideA, w.p, w.kpad, bias, be, g, m, v, M, w.co,
            outS, strideS, outF, strideF, pool);
        break;
      case EPI_MAXPOOL:
        bilinear_wmma_kernel<EPI_MAXPOOL, true><<<grid, blk, 0, stream>>>(
            A, strideA, w.p, w.kpad, bias, be, g, m, v, M, w.co,
            outS, strideS, outF, strideF, pool);
        break;
      default:
        bilinear_wmma_kernel<EPI_RAW, false><<<grid, blk, 0, stream>>>(
            A, strideA, w.p, w.kpad, bias, be, g, m, v, M, w.co,
            outS, strideS, outF, strideF, pool);
        break;
    }
  };

  fill_u32<<<(3 * 16 * 1024 + 255) / 256, 256, 0, stream>>>(pools, 0u, 3 * 16 * 1024);
  build_sign_x9<<<(PTS + 255) / 256, 256, 0, stream>>>(pos, feat, R1);

  gemm(R1, 64,  bw[0], 93, 72, PTS, EPI_SIGN, R2, 64, nullptr, 0, nullptr);
  gemm(R2, 64,  bw[1], 95, 76, PTS, EPI_SIGN, R1, 128, nullptr, 0, nullptr);
  gemm(R1, 128, bw[2], 97, 80, PTS, EPI_MAXPOOL, nullptr, 0, nullptr, 0, pool1);
  pool_sign_kernel<<<(16384 + 255) / 256, 256, 0, stream>>>(pool1, fcA, 16384);
  gemm(fcA, 1024, bw[3], 99, 84, NBATCH, EPI_SIGN, fcB, 512, nullptr, 0, nullptr);
  gemm(fcB, 512,  bw[4], 101, 88, NBATCH, EPI_SIGN, fcA, 256, nullptr, 0, nullptr);
  gemm(fcA, 256,  bw[5], 103, -1, NBATCH, EPI_RAW, nullptr, 0, trans, 9, nullptr);
  add_eye_kernel<<<1, 64, 0, stream>>>(trans, 3);

  einsum_xyz_kernel<<<(PTS + 255) / 256, 256, 0, stream>>>(pos, feat, trans, R1);

  gemm(R1, 64, bw[12], 35, 22, PTS, EPI_BOTH, R2, 64, F64, 64, nullptr);

  gemm(R2, 64,  bw[6], 61, 40, PTS, EPI_SIGN, R1, 64, nullptr, 0, nullptr);
  gemm(R1, 64,  bw[7], 63, 44, PTS, EPI_SIGN, R2, 128, nullptr, 0, nullptr);
  gemm(R2, 128, bw[8], 65, 48, PTS, EPI_MAXPOOL, nullptr, 0, nullptr, 0, pool2);
  pool_sign_kernel<<<(16384 + 255) / 256, 256, 0, stream>>>(pool2, fcA, 16384);
  gemm(fcA, 1024, bw[9],  67, 52, NBATCH, EPI_SIGN, fcB, 512, nullptr, 0, nullptr);
  gemm(fcB, 512,  bw[10], 69, 56, NBATCH, EPI_SIGN, fcA, 256, nullptr, 0, nullptr);
  gemm(fcA, 256,  bw[11], 71, -1, NBATCH, EPI_RAW, nullptr, 0, tfeat, 4096, nullptr);
  add_eye_kernel<<<(16 * 64 + 255) / 256, 256, 0, stream>>>(tfeat, 64);

  einsum_feat_kernel<<<PTS / 4, 256, 0, stream>>>(F64, tfeat, HF);

  gemm(HF + 1024, 1088, bw[13], 37, 26, PTS, EPI_SIGN, R1, 128, nullptr, 0, nullptr);
  gemm(R1, 128,         bw[14], 39, 30, PTS, EPI_MAXPOOL, nullptr, 0, nullptr, 0, pool3);
  pool_sign_kernel<<<(16384 + 255) / 256, 256, 0, stream>>>(pool3, sp3, 16384);
  bcast_global_sign<<<PTS, 256, 0, stream>>>(sp3, HF);

  gemm(HF, 1088, bw[15], 15, 2,  PTS, EPI_SIGN, R1, 512, nullptr, 0, nullptr);
  gemm(R1, 512,  bw[16], 17, 6,  PTS, EPI_SIGN, R2, 256, nullptr, 0, nullptr);
  gemm(R2, 256,  bw[17], 19, 10, PTS, EPI_FLOAT_HT, nullptr, 0, H3F, 128, nullptr);
  head_final_kernel<<<PTS / 256, 256, 0, stream>>>(
      H3F, (const float*)d_in[20], (const float*)d_in[21], out);
}
